// MultiHeadAttention_678604832830
// MI455X (gfx1250) — compile-verified, matrix operand order checked by reading, not run on hardware
//
#include <hip/hip_runtime.h>

// ---------------------------------------------------------------------------
// MultiHeadAttention forward for MI455X (gfx1250), wave32 + WMMA.
// D_MODEL=1024, N_HEAD=16, D_K=D_V=64, B=8, L=1024
// Outputs (concat, f32): ln (8M) | attn_flat (128M) | one_head (8M)
// ---------------------------------------------------------------------------

#define DM 1024
#define NH 16
#define DK 64
#define BB 8
#define LL 1024

typedef __attribute__((ext_vector_type(16))) _Float16 v16h;
typedef __attribute__((ext_vector_type(8)))  _Float16 v8h;
typedef __attribute__((ext_vector_type(8)))  float    v8f;
typedef int v4i __attribute__((vector_size(16)));   // GCC-style, matches builtin proto

union Frag { v16h v; v8h h2[2]; };

#define WMMA_F16(A, Bf, C) \
  __builtin_amdgcn_wmma_f32_16x16x32_f16(false, (A), false, (Bf), (short)0, (C), false, false)

// gfx1250 async global->LDS copy path (ASYNCcnt), with portable fallback.
#if defined(__has_builtin)
#if __has_builtin(__builtin_amdgcn_global_load_async_to_lds_b128) && \
    __has_builtin(__builtin_amdgcn_s_wait_asynccnt)
#define HAS_ASYNC_LDS 1
#endif
#endif
#ifndef HAS_ASYNC_LDS
#define HAS_ASYNC_LDS 0
#endif

__device__ __forceinline__ v8f zero8() {
  v8f z = {0.f, 0.f, 0.f, 0.f, 0.f, 0.f, 0.f, 0.f};
  return z;
}

static constexpr size_t OUT_ATTN = (size_t)BB * LL * DM;                 // 8M floats
static constexpr size_t OUT_ONEH = OUT_ATTN + (size_t)NH * BB * LL * LL; // +128M floats

// ---------------------------------------------------------------------------
// Weight prep: transpose+convert w_q/w_k/w_v -> [h][n][k] f16, proj_w -> f16.
// ---------------------------------------------------------------------------
__global__ void prep_weights(const float* __restrict__ wq, const float* __restrict__ wk,
                             const float* __restrict__ wv, const float* __restrict__ pw,
                             _Float16* __restrict__ wqT, _Float16* __restrict__ wkT,
                             _Float16* __restrict__ wvT, _Float16* __restrict__ pwh) {
  int idx = blockIdx.x * blockDim.x + threadIdx.x;   // 4M total
  int which = idx >> 20;
  int i = idx & ((1 << 20) - 1);
  if (which < 3) {
    int h = i >> 16;            // [h][n][k] flat: h*65536 + n*1024 + k
    int n = (i >> 10) & 63;
    int kk = i & 1023;
    const float* src = (which == 0) ? wq : (which == 1) ? wk : wv;
    _Float16* dst = (which == 0) ? wqT : (which == 1) ? wkT : wvT;
    dst[i] = (_Float16)src[((size_t)h * DM + kk) * DK + n];
  } else {
    pwh[i] = (_Float16)pw[i];
  }
}

// ---------------------------------------------------------------------------
// QKV projection: q_s/k_s [h][b][l][d] f16, v stored transposed [h][b][d][l].
// grid(64, 128, 3), block 128 (4 waves; wave w -> n-tile w of 4).
// ---------------------------------------------------------------------------
__global__ void __launch_bounds__(128) qkv_proj(
    const float* __restrict__ q, const float* __restrict__ k, const float* __restrict__ v,
    const _Float16* __restrict__ wqT, const _Float16* __restrict__ wkT,
    const _Float16* __restrict__ wvT,
    _Float16* __restrict__ qs, _Float16* __restrict__ ks, _Float16* __restrict__ vst) {
  const int z = blockIdx.z;
  const float* X = (z == 0) ? q : (z == 1) ? k : v;
  const _Float16* W = (z == 0) ? wqT : (z == 1) ? wkT : wvT;
  const int hb = blockIdx.y;                  // h*8 + b
  const int h = hb >> 3, b = hb & 7;
  const int mt = blockIdx.x;
  const int wv_ = threadIdx.x >> 5;
  const int lane = threadIdx.x & 31;
  const int m = lane & 15, hi = lane >> 4, n = lane & 15;

  const float* Xrow = X + ((size_t)b * LL + (mt * 16 + m)) * DM;
  const _Float16* Wrow = W + ((size_t)h * DK + (wv_ * 16 + n)) * DM;

  v8f acc = zero8();
  for (int kc = 0; kc < DM; kc += 32) {
    __builtin_prefetch(Xrow + kc + 256, 0, 0);
    Frag a, bf;
    v8f lo = *(const v8f*)(Xrow + kc + hi * 8);
    v8f hh = *(const v8f*)(Xrow + kc + 16 + hi * 8);
    a.h2[0] = __builtin_convertvector(lo, v8h);
    a.h2[1] = __builtin_convertvector(hh, v8h);
    bf.h2[0] = *(const v8h*)(Wrow + kc + hi * 8);
    bf.h2[1] = *(const v8h*)(Wrow + kc + 16 + hi * 8);
    acc = WMMA_F16(a.v, bf.v, acc);
  }

  const int col = wv_ * 16 + n;
  if (z != 2) {
    _Float16* dst = ((z == 0) ? qs : ks) + ((size_t)hb * LL) * DK;
#pragma unroll
    for (int r = 0; r < 8; ++r) {
      int mr = mt * 16 + r + 8 * hi;
      dst[(size_t)mr * DK + col] = (_Float16)acc[r];
    }
  } else {
    _Float16* dst = vst + ((size_t)hb * DK) * LL;   // [d][l]
#pragma unroll
    for (int r = 0; r < 8; ++r) {
      int mr = mt * 16 + r + 8 * hi;
      dst[(size_t)col * LL + mr] = (_Float16)acc[r];
    }
  }
}

// ---------------------------------------------------------------------------
// Fused attention: scores -> mask/scale -> softmax -> attn (f32 out + f16 LDS)
// -> attn @ V -> concat f16. grid(64, 128), block 256, dyn LDS 96KB.
// ---------------------------------------------------------------------------
__global__ void __launch_bounds__(256) attn_kernel(
    const _Float16* __restrict__ qs, const _Float16* __restrict__ ks,
    const _Float16* __restrict__ vst, const unsigned char* __restrict__ mask,
    float* __restrict__ out, _Float16* __restrict__ cc) {
  extern __shared__ char smem[];
  float* sc = (float*)smem;                                     // [16][1024] f32
  _Float16* at = (_Float16*)(smem + 16 * 1024 * sizeof(float)); // [16][1024] f16
  __shared__ float red[16][17];
  __shared__ float rowmax[16], rowsum[16];

  const int hb = blockIdx.y;
  const int h = hb >> 3, b = hb & 7;
  const int mt = blockIdx.x;
  const int wv_ = threadIdx.x >> 5;
  const int lane = threadIdx.x & 31;
  const int m = lane & 15, hi = lane >> 4, n = lane & 15;

  const _Float16* Q = qs + ((size_t)hb * LL) * DK;
  const _Float16* K = ks + ((size_t)hb * LL) * DK;
  const _Float16* Vt = vst + ((size_t)hb * DK) * LL;
  const _Float16* Qrow = Q + (size_t)(mt * 16 + m) * DK;
  const float scale = 0.03125f;  // 1/sqrt(D_MODEL) = 1/32

  // Phase 1: scores. Q fragment is j-invariant: hoist into registers.
  Frag a0, a1;
  a0.h2[0] = *(const v8h*)(Qrow + 0 + hi * 8);
  a0.h2[1] = *(const v8h*)(Qrow + 16 + hi * 8);
  a1.h2[0] = *(const v8h*)(Qrow + 32 + hi * 8);
  a1.h2[1] = *(const v8h*)(Qrow + 48 + hi * 8);

  for (int j = 0; j < 8; ++j) {
    const int kb = (wv_ * 8 + j) * 16;
    const _Float16* Krow = K + (size_t)(kb + n) * DK;
    Frag b0, b1;
    b0.h2[0] = *(const v8h*)(Krow + 0 + hi * 8);
    b0.h2[1] = *(const v8h*)(Krow + 16 + hi * 8);
    b1.h2[0] = *(const v8h*)(Krow + 32 + hi * 8);
    b1.h2[1] = *(const v8h*)(Krow + 48 + hi * 8);
    v8f acc = zero8();
    acc = WMMA_F16(a0.v, b0.v, acc);
    acc = WMMA_F16(a1.v, b1.v, acc);

    const unsigned char* mrow = mask + ((size_t)b * LL + mt * 16) * LL + kb + n;
#pragma unroll
    for (int r = 0; r < 8; ++r) {
      int rr = r + 8 * hi;
      float s = acc[r] * scale;
      if (mrow[(size_t)rr * LL]) s = -__builtin_huge_valf();
      sc[rr * 1024 + kb + n] = s;
    }
  }
  __syncthreads();

  // Phase 2: softmax over each of 16 rows (16 threads/row, 64 cols each)
  const int row = threadIdx.x >> 4;
  const int sub = threadIdx.x & 15;
  float* srow = sc + row * 1024;
  float mx = -__builtin_huge_valf();
  for (int c = sub; c < 1024; c += 16) mx = fmaxf(mx, srow[c]);
  red[row][sub] = mx;
  __syncthreads();
  if (sub == 0) {
    float t = red[row][0];
    for (int i = 1; i < 16; ++i) t = fmaxf(t, red[row][i]);
    rowmax[row] = t;
  }
  __syncthreads();
  const float rm = rowmax[row];
  float sm = 0.f;
  for (int c = sub; c < 1024; c += 16) {
    float e = __expf(srow[c] - rm);
    srow[c] = e;
    sm += e;
  }
  red[row][sub] = sm;
  __syncthreads();
  if (sub == 0) {
    float t = 0.f;
    for (int i = 0; i < 16; ++i) t += red[row][i];
    rowsum[row] = t;
  }
  __syncthreads();
  const float inv = 1.0f / rowsum[row];
  float* arow = out + OUT_ATTN + ((size_t)hb * LL + (mt * 16 + row)) * LL;
  for (int c = sub; c < 1024; c += 16) {
    float p = srow[c] * inv;
    arow[c] = p;
    at[row * 1024 + c] = (_Float16)p;
  }
  __syncthreads();

  // Phase 3: out = attn(16x1024) @ V(1024x64); waves 0..3 take 16-col tiles
  if (wv_ < 4) {
    const _Float16* Vrow = Vt + (size_t)(wv_ * 16 + n) * LL;  // contiguous over keys
    const _Float16* Arow = at + m * 1024;
    v8f acc = zero8();
    for (int kc = 0; kc < LL; kc += 32) {
      __builtin_prefetch(Vrow + kc + 256, 0, 0);
      Frag a, bf;
      a.h2[0] = *(const v8h*)(Arow + kc + hi * 8);
      a.h2[1] = *(const v8h*)(Arow + kc + 16 + hi * 8);
      bf.h2[0] = *(const v8h*)(Vrow + kc + hi * 8);
      bf.h2[1] = *(const v8h*)(Vrow + kc + 16 + hi * 8);
      acc = WMMA_F16(a.v, bf.v, acc);
    }
    _Float16* ccb = cc + ((size_t)b * LL + mt * 16) * DM + h * DK + wv_ * 16 + n;
#pragma unroll
    for (int r = 0; r < 8; ++r) ccb[(size_t)(r + 8 * hi) * DM] = (_Float16)acc[r];
  }
}

// ---------------------------------------------------------------------------
// Output projection + bias + residual + LayerNorm.
// grid(512), block 256 (8 waves x 8 n-tiles = full 1024 cols), dyn LDS 96KB:
//   rb    f32[16][1024] @ 0      (64KB)  row buffer for LN
//   atile f16[16][1024] @ 64KB   (32KB)  async-staged A tile (if available)
// ---------------------------------------------------------------------------
__global__ void __launch_bounds__(256) proj_ln_kernel(
    const _Float16* __restrict__ cc, const _Float16* __restrict__ pwh,
    const float* __restrict__ pb, const float* __restrict__ qin,
    const float* __restrict__ lng, const float* __restrict__ lnb,
    float* __restrict__ out) {
  extern __shared__ float rb[];  // [16][1024] f32
  _Float16* atile = (_Float16*)((char*)rb + 64 * 1024);
  __shared__ float red[16][17];
  __shared__ float rmean[16], rvar[16];

  const int mt = blockIdx.x;     // tile over B*L
  const int wv_ = threadIdx.x >> 5;
  const int lane = threadIdx.x & 31;
  const int m = lane & 15, hi = lane >> 4, n = lane & 15;

#if HAS_ASYNC_LDS
  // Stage the 16x1024 f16 A-tile once via the async DMA path (ASYNCcnt),
  // instead of 8 waves each re-reading it from global.
  {
    const _Float16* Abase = cc + (size_t)(mt * 16) * DM;
    for (int i = threadIdx.x; i < 2048; i += 256) {   // 2048 x 16B = 32KB
      __builtin_amdgcn_global_load_async_to_lds_b128(
          (__attribute__((address_space(1))) v4i*)(Abase + (size_t)i * 8),
          (__attribute__((address_space(3))) v4i*)(atile + (size_t)i * 8),
          0, 0);
    }
    __builtin_amdgcn_s_wait_asynccnt(0);
    __syncthreads();
  }
  const _Float16* Arow = atile + m * 1024;
#else
  const _Float16* Arow = cc + (size_t)(mt * 16 + m) * DM;
#endif

  v8f acc[8];
#pragma unroll
  for (int j = 0; j < 8; ++j) acc[j] = zero8();

  for (int kc = 0; kc < DM; kc += 32) {
    Frag a;
    a.h2[0] = *(const v8h*)(Arow + kc + hi * 8);
    a.h2[1] = *(const v8h*)(Arow + kc + 16 + hi * 8);
#pragma unroll
    for (int j = 0; j < 8; ++j) {
      const int col = (wv_ * 8 + j) * 16 + n;
      const _Float16* Wr = pwh + (size_t)col * DM;   // out[m][col] = sum_k cc[m][k]*pw[col][k]
      __builtin_prefetch(Wr + kc + 256, 0, 0);
      Frag bf;
      bf.h2[0] = *(const v8h*)(Wr + kc + hi * 8);
      bf.h2[1] = *(const v8h*)(Wr + kc + 16 + hi * 8);
      acc[j] = WMMA_F16(a.v, bf.v, acc[j]);
    }
  }

  // epilogue: + bias + residual(q) -> LDS row buffer
#pragma unroll
  for (int j = 0; j < 8; ++j) {
    const int col = (wv_ * 8 + j) * 16 + n;
    const float bias = pb[col];
#pragma unroll
    for (int r = 0; r < 8; ++r) {
      int rr = r + 8 * hi;
      int Mg = mt * 16 + rr;  // global row over B*L
      rb[rr * 1024 + col] = acc[j][r] + bias + qin[(size_t)Mg * DM + col];
    }
  }
  __syncthreads();

  // LayerNorm (population variance), 16 threads per row
  const int row = threadIdx.x >> 4;
  const int sub = threadIdx.x & 15;
  float* rrow = rb + row * 1024;
  float s = 0.f, ss = 0.f;
  for (int c = sub; c < 1024; c += 16) {
    float x = rrow[c];
    s += x;
    ss += x * x;
  }
  red[row][sub] = s;
  __syncthreads();
  if (sub == 0) {
    float t = 0.f;
    for (int i = 0; i < 16; ++i) t += red[row][i];
    rmean[row] = t * (1.0f / 1024.0f);
  }
  __syncthreads();
  red[row][sub] = ss;
  __syncthreads();
  if (sub == 0) {
    float t = 0.f;
    for (int i = 0; i < 16; ++i) t += red[row][i];
    float mu = rmean[row];
    rvar[row] = t * (1.0f / 1024.0f) - mu * mu;
  }
  __syncthreads();
  const float mu = rmean[row];
  const float rstd = rsqrtf(rvar[row] + 1e-6f);
  float* orow = out + (size_t)(mt * 16 + row) * DM;
  for (int c = sub; c < 1024; c += 16) {
    orow[c] = (rrow[c] - mu) * rstd * lng[c] + lnb[c];
  }
}

// ---------------------------------------------------------------------------
// one_head_attn[b] = attn_flat[16*b]  (reference's reshape quirk). float4 copy.
// ---------------------------------------------------------------------------
__global__ void one_head_copy(float* __restrict__ out) {
  size_t idx = (size_t)blockIdx.x * blockDim.x + threadIdx.x;  // 2M float4
  size_t b = idx >> 18;                      // 2^18 float4 per (L*L) slab
  size_t r = idx & (((size_t)1 << 18) - 1);
  const size_t slab = (size_t)LL * LL / 4;
  const float4* src = (const float4*)(out + OUT_ATTN) + (b * 16) * slab + r;
  float4* dst = (float4*)(out + OUT_ONEH) + b * slab + r;
  *dst = *src;
}

// ---------------------------------------------------------------------------
extern "C" void kernel_launch(void* const* d_in, const int* in_sizes, int n_in,
                              void* d_out, int out_size, void* d_ws, size_t ws_size,
                              hipStream_t stream) {
  const float* q = (const float*)d_in[0];
  const float* k = (const float*)d_in[1];
  const float* v = (const float*)d_in[2];
  const unsigned char* mask = (const unsigned char*)d_in[3];
  const float* wq = (const float*)d_in[4];
  const float* wk = (const float*)d_in[5];
  const float* wv = (const float*)d_in[6];
  const float* pw = (const float*)d_in[7];
  const float* pb = (const float*)d_in[8];
  const float* g  = (const float*)d_in[9];
  const float* be = (const float*)d_in[10];
  float* out = (float*)d_out;

  _Float16* ws = (_Float16*)d_ws;
  _Float16* wqT = ws;                         // 1M halfs
  _Float16* wkT = wqT + (1 << 20);
  _Float16* wvT = wkT + (1 << 20);
  _Float16* pwh = wvT + (1 << 20);
  _Float16* qs  = pwh + (1 << 20);            // 8M halfs each
  _Float16* ks  = qs  + (8 << 20);
  _Float16* vst = ks  + (8 << 20);
  _Float16* cc  = vst + (8 << 20);

  prep_weights<<<(4 << 20) / 256, 256, 0, stream>>>(wq, wk, wv, pw, wqT, wkT, wvT, pwh);
  qkv_proj<<<dim3(LL / 16, NH * BB, 3), dim3(128), 0, stream>>>(q, k, v, wqT, wkT, wvT,
                                                                qs, ks, vst);
  attn_kernel<<<dim3(LL / 16, NH * BB), dim3(256), 96 * 1024, stream>>>(qs, ks, vst, mask,
                                                                        out, cc);
  // always reserve 96KB so host/device agreement doesn't depend on HAS_ASYNC_LDS
  proj_ln_kernel<<<dim3((BB * LL) / 16), dim3(256), 96 * 1024, stream>>>(cc, pwh, pb, q, g,
                                                                         be, out);
  one_head_copy<<<dim3(8192), dim3(256), 0, stream>>>(out);
}
